// PowerCoderAttention_15547781611674
// MI455X (gfx1250) — matrix operand order repaired and checked by reading, not compiled, not run on hardware
//
#include <hip/hip_runtime.h>

// ---------------- problem constants ----------------
#define T_    2048
#define HID_  2048
#define HQ_   16
#define HKV_  8
#define D_    128
#define SCALE_ 0.08838834764831845f   // D^-0.5
#define LOG_THETA_ 9.210340371976184f // ln(10000)

typedef __bf16 bf16_t;
typedef __bf16 v8bf  __attribute__((ext_vector_type(8)));
typedef __bf16 v16bf __attribute__((ext_vector_type(16)));
typedef float  v8f   __attribute__((ext_vector_type(8)));

__device__ __forceinline__ v16bf combine16(v8bf lo, v8bf hi) {
  return __builtin_shufflevector(lo, hi, 0,1,2,3,4,5,6,7,8,9,10,11,12,13,14,15);
}
__device__ __forceinline__ v8f wmma_bf16(v16bf a, v16bf b, v8f c) {
  return __builtin_amdgcn_wmma_f32_16x16x32_bf16(false, a, false, b, (short)0, c, false, false);
}

// ---------------- conversion kernels ----------------
__global__ void cvt_bf16_kernel(const float* __restrict__ in, bf16_t* __restrict__ out, int n) {
  for (int i = blockIdx.x * blockDim.x + threadIdx.x; i < n; i += gridDim.x * blockDim.x)
    out[i] = (bf16_t)in[i];
}

// in: [K][N] fp32 row-major  ->  out: [N][K] bf16 row-major
__global__ void cvt_transpose_bf16_kernel(const float* __restrict__ in, bf16_t* __restrict__ out,
                                          int K, int N) {
  int total = K * N;
  for (int i = blockIdx.x * blockDim.x + threadIdx.x; i < total; i += gridDim.x * blockDim.x) {
    int k = i / N, c = i - k * N;
    out[(size_t)c * K + k] = (bf16_t)in[i];
  }
}

// NeoX RoPE on fp32 [T][H][128] -> bf16 same layout
__global__ void rope_cvt_kernel(const float* __restrict__ x, const int* __restrict__ pos,
                                bf16_t* __restrict__ out, int H) {
  int total = T_ * H * 64;
  for (int i = blockIdx.x * blockDim.x + threadIdx.x; i < total; i += gridDim.x * blockDim.x) {
    int ii = i & 63;
    int h  = (i >> 6) % H;
    int t  = i / (64 * H);
    float inv = __expf(-((float)(2 * ii) / (float)D_) * LOG_THETA_);
    float f = (float)pos[t] * inv;
    float c = __cosf(f), s = __sinf(f);
    size_t base = ((size_t)t * H + h) * D_;
    float x1 = x[base + ii], x2 = x[base + ii + 64];
    out[base + ii]      = (bf16_t)(x1 * c - x2 * s);
    out[base + ii + 64] = (bf16_t)(x2 * c + x1 * s);
  }
}

// v fp32 [T][HKV*D] -> vt bf16 [HKV*D][T]  (so W·V B-fragments are contiguous)
__global__ void vt_cvt_kernel(const float* __restrict__ v, bf16_t* __restrict__ vt) {
  int total = T_ * HKV_ * D_;
  for (int i = blockIdx.x * blockDim.x + threadIdx.x; i < total; i += gridDim.x * blockDim.x) {
    int t = i >> 10;        // HKV*D = 1024
    int r = i & 1023;       // h*128 + d
    vt[(size_t)r * T_ + t] = (bf16_t)v[i];
  }
}

// ---------------- bf16 WMMA GEMM, 64x64 register tile per wave ----------------
// C[MxN] = A[MxK] * Bt[NxK]^T + bias ; M,N multiples of 64, K multiple of 32
__global__ void __launch_bounds__(256)
gemm_bf16_wmma(const bf16_t* __restrict__ A, const bf16_t* __restrict__ Bt,
               const float* __restrict__ bias, float* __restrict__ C,
               int M, int N, int K) {
  const int lane  = threadIdx.x & 31;
  const int wid   = blockIdx.x * (blockDim.x >> 5) + (threadIdx.x >> 5);
  const int tiles_n = N >> 6;                 // 64-wide macro tiles
  const int tiles_m = M >> 6;
  if (wid >= tiles_m * tiles_n) return;
  const int tm = wid / tiles_n, tn = wid - tm * tiles_n;
  const int ln    = lane & 15;
  const int hi16  = (lane >> 4) & 1;
  const int abase = hi16 ? 16 : 0;  // A-fragment K sub-offset (contiguous k = 16h + s)
  const int mbase = hi16 ? 8 : 0;   // C layout M offset

  const bf16_t* arow[4];
  const bf16_t* brow[4];
#pragma unroll
  for (int mi = 0; mi < 4; ++mi)
    arow[mi] = A + (size_t)(tm * 64 + mi * 16 + ln) * K + abase;
#pragma unroll
  for (int ni = 0; ni < 4; ++ni)
    brow[ni] = Bt + (size_t)(tn * 64 + ni * 16 + ln) * K + (hi16 ? 16 : 0);

  v8f acc[4][4];
#pragma unroll
  for (int mi = 0; mi < 4; ++mi)
#pragma unroll
    for (int ni = 0; ni < 4; ++ni) acc[mi][ni] = (v8f){};

  for (int kc = 0; kc < K; kc += 32) {
    // hint K data 512 elements (16 K steps) ahead into cache (global_prefetch_b8)
    __builtin_prefetch(arow[0] + kc + 512, 0, 1);
    __builtin_prefetch(brow[0] + kc + 512, 0, 1);
    v16bf af[4], bf[4];
#pragma unroll
    for (int mi = 0; mi < 4; ++mi) {
      v8bf lo = *(const v8bf*)(arow[mi] + kc);
      v8bf hi = *(const v8bf*)(arow[mi] + kc + 8);
      af[mi] = combine16(lo, hi);
    }
#pragma unroll
    for (int ni = 0; ni < 4; ++ni)
      bf[ni] = *(const v16bf*)(brow[ni] + kc);
#pragma unroll
    for (int mi = 0; mi < 4; ++mi)
#pragma unroll
      for (int ni = 0; ni < 4; ++ni)
        acc[mi][ni] = wmma_bf16(af[mi], bf[ni], acc[mi][ni]);
  }

#pragma unroll
  for (int ni = 0; ni < 4; ++ni) {
    float bn = bias ? bias[tn * 64 + ni * 16 + ln] : 0.f;
#pragma unroll
    for (int mi = 0; mi < 4; ++mi)
#pragma unroll
      for (int i = 0; i < 8; ++i)
        C[(size_t)(tm * 64 + mi * 16 + i + mbase) * N + tn * 64 + ni * 16 + ln] =
            acc[mi][ni][i] + bn;
  }
}

// ---------------- gating: g = logsigmoid(hs @ Wg + bg), one wave per (t,h) ----------------
__global__ void __launch_bounds__(256)
gate_kernel(const float* __restrict__ hs, const float* __restrict__ Wg,
            const float* __restrict__ bg, float* __restrict__ gl) {
  const int lane = threadIdx.x & 31;
  const int wid  = blockIdx.x * (blockDim.x >> 5) + (threadIdx.x >> 5);
  if (wid >= T_ * HKV_) return;
  const int t = wid / HKV_, h = wid - t * HKV_;
  float sum = 0.f;
  for (int c = lane; c < HID_; c += 32)
    sum += hs[(size_t)t * HID_ + c] * Wg[(size_t)c * HKV_ + h];
  for (int off = 16; off; off >>= 1) sum += __shfl_xor(sum, off);
  if (lane == 0) {
    float x = sum + bg[h];
    gl[(size_t)t * HKV_ + h] = fminf(x, 0.f) - log1pf(__expf(-fabsf(x)));
  }
}

// ---------------- cumsum over time per kv head: G[t][h] = sum_{s<=t} gl[s][h] ----------------
__global__ void __launch_bounds__(256)
cumsum_kernel(const float* __restrict__ gl, float* __restrict__ G) {
  const int h = blockIdx.x;     // 0..HKV-1
  const int tid = threadIdx.x;  // 0..255, each owns 8 consecutive t
  __shared__ float part[256];
  float vals[8]; float s = 0.f;
  for (int j = 0; j < 8; ++j) { vals[j] = gl[(size_t)(tid * 8 + j) * HKV_ + h]; s += vals[j]; }
  part[tid] = s;
  __syncthreads();
  for (int off = 1; off < 256; off <<= 1) {
    float v = (tid >= off) ? part[tid - off] : 0.f;
    __syncthreads();
    part[tid] += v;
    __syncthreads();
  }
  float run = (tid > 0) ? part[tid - 1] : 0.f;
  for (int j = 0; j < 8; ++j) {
    run += vals[j];
    G[(size_t)(tid * 8 + j) * HKV_ + h] = run;
  }
}

// ---------------- attention: one wave per (q-head, 16-row query tile) ----------------
__global__ void __launch_bounds__(128)
attn_wmma(const bf16_t* __restrict__ qb, const bf16_t* __restrict__ kbuf,
          const bf16_t* __restrict__ vt, const float* __restrict__ G,
          bf16_t* __restrict__ attnb) {
  __shared__ __align__(32) bf16_t wlds[4][16 * 32];  // per-wave W staging (16x32 bf16)
  const int lane  = threadIdx.x & 31;
  const int wslot = threadIdx.x >> 5;
  const int wid   = blockIdx.x * 4 + wslot;
  const int h   = wid >> 7;      // 0..15 query head
  const int qt  = wid & 127;     // 0..127 query tile of 16 rows
  const int hkv = h >> 1;        // GQA rep = 2
  const int ln    = lane & 15;
  const int hi16  = (lane >> 4) & 1;
  const int abase = hi16 ? 16 : 0;
  const int mbase = hi16 ? 8 : 0;

  // Q fragments for this tile: 16x128 split into four 16x32 K-chunks
  v16bf qa[4];
  const bf16_t* qrow = qb + (size_t)(qt * 16 + ln) * (HQ_ * D_) + h * D_;
#pragma unroll
  for (int c = 0; c < 4; ++c) {
    v8bf lo = *(const v8bf*)(qrow + c * 32 + abase);
    v8bf hi = *(const v8bf*)(qrow + c * 32 + abase + 8);
    qa[c] = combine16(lo, hi);
  }
  float gq[8];
#pragma unroll
  for (int i = 0; i < 8; ++i)
    gq[i] = G[(size_t)(qt * 16 + i + mbase) * HKV_ + hkv];

  v8f o[8];
#pragma unroll
  for (int dt = 0; dt < 8; ++dt) o[dt] = (v8f){};

  bf16_t* wl = wlds[wslot];
  const int kbmax = qt >> 1;     // causal: key blocks of 32
  for (int kb = 0; kb <= kbmax; ++kb) {
    const int s0 = kb * 32;
    // ---- S = q k^T : two 16x16 tiles over a 32-key block ----
    v8f s0acc = {}; v8f s1acc = {};
    const bf16_t* kp = kbuf + (size_t)(s0 + ln) * (HKV_ * D_) + hkv * D_ + (hi16 ? 16 : 0);
#pragma unroll
    for (int c = 0; c < 4; ++c) {
      v16bf b0 = *(const v16bf*)(kp + c * 32);
      v16bf b1 = *(const v16bf*)(kp + (size_t)16 * (HKV_ * D_) + c * 32);
      s0acc = wmma_bf16(qa[c], b0, s0acc);
      s1acc = wmma_bf16(qa[c], b1, s1acc);
    }
    // ---- elementwise: scale, square (P=2), decay exp(Gq-Gk), causal mask ----
    float gk0 = G[(size_t)(s0 + ln) * HKV_ + hkv];
    float gk1 = G[(size_t)(s0 + 16 + ln) * HKV_ + hkv];
#pragma unroll
    for (int i = 0; i < 8; ++i) {
      int qrowi = qt * 16 + i + mbase;
      float x0 = s0acc[i] * SCALE_; x0 *= x0;
      float w0 = (s0 + ln <= qrowi) ? x0 * __expf(gq[i] - gk0) : 0.f;
      float x1 = s1acc[i] * SCALE_; x1 *= x1;
      float w1 = (s0 + 16 + ln <= qrowi) ? x1 * __expf(gq[i] - gk1) : 0.f;
      wl[(i + mbase) * 32 + ln]      = (bf16_t)w0;
      wl[(i + mbase) * 32 + 16 + ln] = (bf16_t)w1;
    }
    asm volatile("s_wait_dscnt 0" ::: "memory");  // LDS RAW: repack W into A-fragment
    v8bf wlo = *(const v8bf*)(wl + ln * 32 + abase);
    v8bf whi = *(const v8bf*)(wl + ln * 32 + abase + 8);
    v16bf wa = combine16(wlo, whi);
    // ---- O += W(16x32) * V(32x128), v stored transposed [h*D+d][t] ----
    const bf16_t* vp = vt + (size_t)(hkv * D_ + ln) * T_ + s0 + (hi16 ? 16 : 0);
#pragma unroll
    for (int dt = 0; dt < 8; ++dt) {
      v16bf vb = *(const v16bf*)(vp + (size_t)dt * 16 * T_);
      o[dt] = wmma_bf16(wa, vb, o[dt]);
    }
  }
  // write attention output tile (bf16, row-major [T][HQ*D]) for the final GEMM
#pragma unroll
  for (int dt = 0; dt < 8; ++dt)
#pragma unroll
    for (int i = 0; i < 8; ++i)
      attnb[(size_t)(qt * 16 + i + mbase) * (HQ_ * D_) + h * D_ + dt * 16 + ln] = (bf16_t)o[dt][i];
}

// ---------------- host launcher ----------------
extern "C" void kernel_launch(void* const* d_in, const int* in_sizes, int n_in,
                              void* d_out, int out_size, void* d_ws, size_t ws_size,
                              hipStream_t stream) {
  const int*   positions = (const int*)d_in[0];
  const float* hs = (const float*)d_in[1];
  const float* Wq = (const float*)d_in[2];  const float* bq = (const float*)d_in[3];
  const float* Wk = (const float*)d_in[4];  const float* bk = (const float*)d_in[5];
  const float* Wv = (const float*)d_in[6];  const float* bv = (const float*)d_in[7];
  const float* Wg = (const float*)d_in[8];  const float* bg = (const float*)d_in[9];
  const float* Wo = (const float*)d_in[10]; const float* bo = (const float*)d_in[11];
  float* out = (float*)d_out;
  (void)in_sizes; (void)n_in; (void)out_size; (void)ws_size;

  // workspace carve (all offsets 256B aligned)
  char* ws = (char*)d_ws;
  size_t off = 0;
  auto carve = [&](size_t bytes) -> void* {
    void* p = ws + off;
    off = (off + bytes + 255) & ~(size_t)255;
    return p;
  };
  bf16_t* hsb  = (bf16_t*)carve((size_t)T_ * HID_ * 2);          // hs bf16
  bf16_t* WqT  = (bf16_t*)carve((size_t)HID_ * HQ_ * D_ * 2);    // [N][K]
  bf16_t* WkT  = (bf16_t*)carve((size_t)HID_ * HKV_ * D_ * 2);
  bf16_t* WvT  = (bf16_t*)carve((size_t)HID_ * HKV_ * D_ * 2);
  bf16_t* WoT  = (bf16_t*)carve((size_t)HQ_ * D_ * HID_ * 2);
  float*  qf   = (float*)carve((size_t)T_ * HQ_ * D_ * 4);       // fp32 q proj (later aliased by attnb)
  float*  kf   = (float*)carve((size_t)T_ * HKV_ * D_ * 4);
  float*  vf   = (float*)carve((size_t)T_ * HKV_ * D_ * 4);
  bf16_t* qbb  = (bf16_t*)carve((size_t)T_ * HQ_ * D_ * 2);      // bf16 q after rope
  bf16_t* kbb  = (bf16_t*)carve((size_t)T_ * HKV_ * D_ * 2);
  bf16_t* vtb  = (bf16_t*)carve((size_t)HKV_ * D_ * T_ * 2);     // v transposed
  float*  gl   = (float*)carve((size_t)T_ * HKV_ * 4);
  float*  Gbuf = (float*)carve((size_t)T_ * HKV_ * 4);
  bf16_t* attnb = (bf16_t*)qf;  // qf is dead after rope; reuse for attention output (bf16)

  const int TPB = 256;
  const int GSB = 2048;  // grid-stride blocks for elementwise kernels

  // 1) conversions / transposes
  cvt_bf16_kernel<<<GSB, TPB, 0, stream>>>(hs, hsb, T_ * HID_);
  cvt_transpose_bf16_kernel<<<GSB, TPB, 0, stream>>>(Wq, WqT, HID_, HQ_ * D_);
  cvt_transpose_bf16_kernel<<<GSB, TPB, 0, stream>>>(Wk, WkT, HID_, HKV_ * D_);
  cvt_transpose_bf16_kernel<<<GSB, TPB, 0, stream>>>(Wv, WvT, HID_, HKV_ * D_);
  cvt_transpose_bf16_kernel<<<GSB, TPB, 0, stream>>>(Wo, WoT, HQ_ * D_, HID_);

  // 2) projections (bf16 WMMA 64x64 macro tiles, f32 accumulate)
  {
    int tiles = (T_ / 64) * ((HQ_ * D_) / 64);   // 32*32 = 1024 waves
    gemm_bf16_wmma<<<tiles / 8, TPB, 0, stream>>>(hsb, WqT, bq, qf, T_, HQ_ * D_, HID_);
  }
  {
    int tiles = (T_ / 64) * ((HKV_ * D_) / 64);  // 32*16 = 512 waves
    gemm_bf16_wmma<<<tiles / 8, TPB, 0, stream>>>(hsb, WkT, bk, kf, T_, HKV_ * D_, HID_);
    gemm_bf16_wmma<<<tiles / 8, TPB, 0, stream>>>(hsb, WvT, bv, vf, T_, HKV_ * D_, HID_);
  }

  // 3) RoPE + bf16 convert; v transpose-convert
  rope_cvt_kernel<<<GSB, TPB, 0, stream>>>(qf, positions, qbb, HQ_);
  rope_cvt_kernel<<<GSB, TPB, 0, stream>>>(kf, positions, kbb, HKV_);
  vt_cvt_kernel<<<GSB, TPB, 0, stream>>>(vf, vtb);

  // 4) gating + cumsum
  gate_kernel<<<(T_ * HKV_) / 8, TPB, 0, stream>>>(hs, Wg, bg, gl);
  cumsum_kernel<<<HKV_, TPB, 0, stream>>>(gl, Gbuf);

  // 5) attention (writes attnb, aliased onto qf)
  attn_wmma<<<(HQ_ * (T_ / 16)) / 4, 128, 0, stream>>>(qbb, kbb, vtb, Gbuf, attnb);

  // 6) output projection -> d_out (fp32)
  {
    int tiles = (T_ / 64) * (HID_ / 64);  // 1024 waves
    gemm_bf16_wmma<<<tiles / 8, TPB, 0, stream>>>(attnb, WoT, bo, out, T_, HID_, HQ_ * D_);
  }
}
